// ContrastLoss_16509854286623
// MI455X (gfx1250) — matrix operand order repaired and checked by reading, not compiled, not run on hardware
//
#include <hip/hip_runtime.h>
#include <hip/hip_bf16.h>
#include <stdint.h>

typedef __attribute__((ext_vector_type(16))) __bf16 v16bf;
typedef __attribute__((ext_vector_type(8)))  float  v8f;

#define C_IN    512
#define HW_     16384
#define N_IMG   8
#define P_TOT   (N_IMG*HW_)      // 131072 pixels
#define FDIM    256
#define TEMP_   0.07f
#define BN_EPS_ 1e-5f

// ---------------------------------------------------------------------------
// Zero the stat scratch (S + colsum + mean + invstd, contiguous) and the output
// ---------------------------------------------------------------------------
__global__ void k_zero(float* ws, int n, float* out) {
  int i = blockIdx.x * blockDim.x + threadIdx.x;
  if (i < n) ws[i] = 0.f;
  if (i == 0) out[0] = 0.f;
}

// ---------------------------------------------------------------------------
// colsum[c] = sum_p X[p,c]   (X[p,c] = preds[n][c][hw], p = n*HW + hw)
// ---------------------------------------------------------------------------
__global__ void k_colsum(const float* __restrict__ preds, float* __restrict__ colsum) {
  __shared__ float red[256];
  const int c = blockIdx.x;
  float s = 0.f;
  for (int n = 0; n < N_IMG; ++n) {
    const float* base = preds + ((size_t)n * C_IN + c) * HW_;
    for (int i = threadIdx.x; i < HW_; i += 256) s += base[i];
  }
  red[threadIdx.x] = s;
  __syncthreads();
  for (int off = 128; off > 0; off >>= 1) {
    if (threadIdx.x < off) red[threadIdx.x] += red[threadIdx.x + off];
    __syncthreads();
  }
  if (threadIdx.x == 0) colsum[c] = red[0];
}

// ---------------------------------------------------------------------------
// Gram: S += X^T X over a chunk of 8192 pixels (one image segment per z-block).
// WG = 8 waves in 2x4 over a 128x128 tile of S; each wave owns a 64x32 fp32
// accumulator (8 WMMA tiles = 64 VGPRs). All loads are base+offset off the
// kernarg pointer so InferAddressSpaces keeps them global_load_b128 (flat
// loads would tick DScnt too). B fragments convert first; each A fragment is
// loaded and immediately consumed so fp32 staging dies early.
// ---------------------------------------------------------------------------
__global__ __launch_bounds__(256, 1)
void k_gram(const float* __restrict__ preds, float* __restrict__ S) {
  const int lane = threadIdx.x & 31;
  const int wave = threadIdx.x >> 5;   // 0..7
  const int half = lane >> 4;          // 0: lanes 0-15, 1: lanes 16-31
  const int m16  = lane & 15;
  const int iBase = blockIdx.x * 128 + (wave >> 2) * 64;   // 2 wave-rows of 64
  const int jBase = blockIdx.y * 128 + (wave & 3) * 32;    // 4 wave-cols of 32
  const int p0  = blockIdx.z * 8192;   // stays within one image (8192 | 16384)
  const int n   = p0 >> 14;
  const int hw0 = p0 & (HW_ - 1);
  const float* img = preds + (size_t)n * C_IN * HW_;

  // Per-lane element offsets (from img), pre-offset by the lane-half K shift.
  // A (16x32, 16-bit) layout: lane half 0 -> K {0..7,16..23}, half 1 -> +8.
  // B (32x16, 16-bit) layout: lane half 0 -> K {0..7,16..23},      half 1 -> +8.
  size_t aBase[4], bBase[2];
#pragma unroll
  for (int mi = 0; mi < 4; ++mi)
    aBase[mi] = (size_t)(iBase + mi * 16 + m16) * HW_ + hw0 + half * 8;
#pragma unroll
  for (int nj = 0; nj < 2; ++nj)
    bBase[nj] = (size_t)(jBase + nj * 16 + m16) * HW_ + hw0 + half * 8;

  v8f acc[4][2];
#pragma unroll
  for (int a = 0; a < 4; ++a)
#pragma unroll
    for (int b = 0; b < 2; ++b)
#pragma unroll
      for (int r = 0; r < 8; ++r) acc[a][b][r] = 0.f;

#pragma unroll 1
  for (int k = 0; k < 8192; k += 32) {
    v16bf bfr[2];
#pragma unroll
    for (int nj = 0; nj < 2; ++nj) {
      const float4* bp = (const float4*)(img + bBase[nj] + k);
#pragma unroll
      for (int q = 0; q < 4; ++q) {
        float4 f = bp[(q < 2) ? q : q + 2];    // floats 0..7,16..23 -> bfr[0..15]
        bfr[nj][4 * q + 0] = (__bf16)f.x;
        bfr[nj][4 * q + 1] = (__bf16)f.y;
        bfr[nj][4 * q + 2] = (__bf16)f.z;
        bfr[nj][4 * q + 3] = (__bf16)f.w;
      }
    }
#pragma unroll
    for (int mi = 0; mi < 4; ++mi) {
      const float4* ap = (const float4*)(img + aBase[mi] + k);
      float4 f0 = ap[0];                       // floats 0..3   -> afr[0..3]
      float4 f1 = ap[1];                       // floats 4..7   -> afr[4..7]
      float4 f2 = ap[4];                       // floats 16..19 -> afr[8..11]
      float4 f3 = ap[5];                       // floats 20..23 -> afr[12..15]
      v16bf afr;
      afr[0]  = (__bf16)f0.x; afr[1]  = (__bf16)f0.y;
      afr[2]  = (__bf16)f0.z; afr[3]  = (__bf16)f0.w;
      afr[4]  = (__bf16)f1.x; afr[5]  = (__bf16)f1.y;
      afr[6]  = (__bf16)f1.z; afr[7]  = (__bf16)f1.w;
      afr[8]  = (__bf16)f2.x; afr[9]  = (__bf16)f2.y;
      afr[10] = (__bf16)f2.z; afr[11] = (__bf16)f2.w;
      afr[12] = (__bf16)f3.x; afr[13] = (__bf16)f3.y;
      afr[14] = (__bf16)f3.z; afr[15] = (__bf16)f3.w;
      acc[mi][0] = __builtin_amdgcn_wmma_f32_16x16x32_bf16(
          false, afr, false, bfr[0], (short)0, acc[mi][0], false, false);
      acc[mi][1] = __builtin_amdgcn_wmma_f32_16x16x32_bf16(
          false, afr, false, bfr[1], (short)0, acc[mi][1], false, false);
    }
  }

  // C/D layout: VGPR r -> M = r + half*8, N = lane&15
#pragma unroll
  for (int mi = 0; mi < 4; ++mi)
#pragma unroll
    for (int nj = 0; nj < 2; ++nj)
#pragma unroll
      for (int r = 0; r < 8; ++r) {
        int row = iBase + mi * 16 + r + half * 8;
        int col = jBase + nj * 16 + m16;
        atomicAdd(&S[(size_t)row * C_IN + col], acc[mi][nj][r]);
      }
}

// ---------------------------------------------------------------------------
// BN statistics from Gram: mean_o = w1_o.colsum/P + b1_o
//                          E[h^2]_o = (w1_o^T S w1_o + 2 b1 w1_o.colsum)/P + b1^2
// ---------------------------------------------------------------------------
__global__ void k_stats(const float* __restrict__ S, const float* __restrict__ colsum,
                        const float* __restrict__ w1, const float* __restrict__ b1,
                        float* __restrict__ meanArr, float* __restrict__ invstdArr) {
  __shared__ float redq[256], redd[256];
  const int o = blockIdx.x;
  const float* wrow = w1 + (size_t)o * C_IN;
  float q = 0.f, d = 0.f;
  for (int c = threadIdx.x; c < C_IN; c += 256) {
    float wc = wrow[c];
    d += wc * colsum[c];
    const float* srow = S + (size_t)c * C_IN;
    float t = 0.f;
    for (int c2 = 0; c2 < C_IN; ++c2) t += srow[c2] * wrow[c2];
    q += wc * t;
  }
  redq[threadIdx.x] = q; redd[threadIdx.x] = d;
  __syncthreads();
  for (int off = 128; off > 0; off >>= 1) {
    if (threadIdx.x < off) {
      redq[threadIdx.x] += redq[threadIdx.x + off];
      redd[threadIdx.x] += redd[threadIdx.x + off];
    }
    __syncthreads();
  }
  if (threadIdx.x == 0) {
    const float invP = 1.f / (float)P_TOT;
    float b    = b1[o];
    float mean = redd[0] * invP + b;
    float ex2  = (redq[0] + 2.f * b * redd[0]) * invP + b * b;
    float var  = ex2 - mean * mean;
    meanArr[o]   = mean;
    invstdArr[o] = rsqrtf(var + BN_EPS_);
  }
}

// ---------------------------------------------------------------------------
// Gather 16 selected pixels, run conv1->BN->ReLU->conv2->L2norm, emit bf16 F.
// cf row r maps to (t = r % T, v = r / T); pixel = pix_idx[t][v], img = batch_idx[t].
// ---------------------------------------------------------------------------
__global__ void k_embed(const float* __restrict__ preds,
                        const float* __restrict__ w1, const float* __restrict__ b1,
                        const float* __restrict__ gamma, const float* __restrict__ beta,
                        const float* __restrict__ w2, const float* __restrict__ b2,
                        const float* __restrict__ meanArr, const float* __restrict__ invstdArr,
                        const int* __restrict__ batch_idx, const int* __restrict__ pix_idx,
                        int T, int n_view, int M,
                        unsigned short* __restrict__ Fb) {
  __shared__ float xsg[16 * 512];   // x-gather, later reused as g[16*256]
  __shared__ float hbuf[16 * 512];
  __shared__ int   bArr[16], pArr[16];
  __shared__ float ps[256];
  __shared__ float invn[16];

  const int tid = threadIdx.x;
  const int r0  = blockIdx.x * 16;

  if (tid < 16) {
    int r = r0 + tid;
    int t = 0, v = 0;
    if (r < M) { t = r % T; v = r / T; }
    bArr[tid] = batch_idx[t];
    pArr[tid] = pix_idx[t * n_view + v];
  }
  __syncthreads();

  for (int idx = tid; idx < 16 * 512; idx += 256) {
    int m = idx >> 9, c = idx & 511;
    xsg[idx] = preds[((size_t)bArr[m] * C_IN + c) * HW_ + pArr[m]];
  }
  __syncthreads();

  // h1 = x.W1^T + b1 -> BN -> ReLU
  for (int idx = tid; idx < 16 * 512; idx += 256) {
    int m = idx >> 9, o = idx & 511;
    const float* wr = w1 + (size_t)o * C_IN;
    const float* xr = xsg + m * 512;
    float acc = 0.f;
    for (int c = 0; c < C_IN; ++c) acc += xr[c] * wr[c];
    acc += b1[o];
    acc = (acc - meanArr[o]) * invstdArr[o] * gamma[o] + beta[o];
    hbuf[idx] = fmaxf(acc, 0.f);
  }
  __syncthreads();

  // h2 = relu(h1).W2^T + b2 (writes alias xsg; xsg fully consumed above)
  for (int idx = tid; idx < 16 * 256; idx += 256) {
    int m = idx >> 8, o = idx & 255;
    const float* wr = w2 + (size_t)o * C_IN;
    const float* hr = hbuf + m * 512;
    float acc = b2[o];
    for (int c = 0; c < C_IN; ++c) acc += hr[c] * wr[c];
    xsg[idx] = acc;
  }
  __syncthreads();

  { // per-row sum of squares
    int m = tid >> 4, l = tid & 15;
    float s = 0.f;
    for (int o = l; o < 256; o += 16) { float v = xsg[m * 256 + o]; s += v * v; }
    ps[tid] = s;
  }
  __syncthreads();
  if (tid < 16) {
    float s = 0.f;
    for (int i = 0; i < 16; ++i) s += ps[tid * 16 + i];
    invn[tid] = 1.f / fmaxf(sqrtf(s), 1e-12f);
  }
  __syncthreads();

  for (int idx = tid; idx < 16 * 256; idx += 256) {
    int m = idx >> 8, o = idx & 255;
    float v = xsg[idx] * invn[m];
    __bf16 hv = (__bf16)v;
    Fb[(size_t)(r0 + m) * FDIM + o] = __builtin_bit_cast(unsigned short, hv);
  }
}

// ---------------------------------------------------------------------------
// logits = F.F^T via WMMA bf16 (one wave per 16x16 tile, K = 256)
// ---------------------------------------------------------------------------
__global__ void k_logits(const unsigned short* __restrict__ FbUS,
                         float* __restrict__ logits, int Mpad) {
  const int lane = threadIdx.x & 31;
  const int half = lane >> 4;
  const int m16  = lane & 15;
  const int i0 = blockIdx.x * 16, j0 = blockIdx.y * 16;
  union U { v16bf v; uint32_t u[8]; };
  v8f acc;
#pragma unroll
  for (int r = 0; r < 8; ++r) acc[r] = 0.f;

  const uint32_t* Fu = (const uint32_t*)FbUS;   // bf16 pairs, row stride 128 u32
  for (int kk = 0; kk < FDIM; kk += 32) {
    U a, b;
    const uint32_t* arow = Fu + (size_t)(i0 + m16) * (FDIM / 2) + kk / 2;
    const uint32_t* brow = Fu + (size_t)(j0 + m16) * (FDIM / 2) + kk / 2;  // B[k][j]=F[j][k]
#pragma unroll
    for (int r = 0; r < 8; ++r) {
      int ao = ((r < 4) ? 2 * r : 8 + 2 * r) + (half ? 8 : 0);
      int bo = ((r < 4) ? 2 * r : 8 + 2 * r) + (half ? 8 : 0);
      a.u[r] = arow[ao / 2];
      b.u[r] = brow[bo / 2];
    }
    acc = __builtin_amdgcn_wmma_f32_16x16x32_bf16(
        false, a.v, false, b.v, (short)0, acc, false, false);
  }
#pragma unroll
  for (int r = 0; r < 8; ++r)
    logits[(size_t)(i0 + r + half * 8) * Mpad + (j0 + m16)] = acc[r];
}

// ---------------------------------------------------------------------------
// Per-row contrastive reduction; atomicAdd scalar loss contribution.
// ---------------------------------------------------------------------------
__global__ void k_loss(const float* __restrict__ logits, const int* __restrict__ y_cls,
                       int T, int M, int Mpad, float* __restrict__ out) {
  extern __shared__ float lrow[];
  __shared__ float red[256];
  __shared__ float red2[256];
  const int i   = blockIdx.x;
  const int tid = threadIdx.x;
  const float invT = 1.f / TEMP_;
  const int labi = y_cls[i % T];

  for (int j = tid; j < M; j += 256) lrow[j] = logits[(size_t)i * Mpad + j] * invT;
  __syncthreads();

  float mx = -3.4e38f;
  for (int j = tid; j < M; j += 256) mx = fmaxf(mx, lrow[j]);
  red[tid] = mx; __syncthreads();
  for (int off = 128; off > 0; off >>= 1) {
    if (tid < off) red[tid] = fmaxf(red[tid], red[tid + off]);
    __syncthreads();
  }
  const float rmax = red[0];
  __syncthreads();

  float pn = 0.f;
  for (int j = tid; j < M; j += 256)
    if (y_cls[j % T] != labi) pn += expf(lrow[j] - rmax);
  red[tid] = pn; __syncthreads();
  for (int off = 128; off > 0; off >>= 1) {
    if (tid < off) red[tid] += red[tid + off];
    __syncthreads();
  }
  const float neg = red[0];
  __syncthreads();

  float psum = 0.f, pcnt = 0.f;
  for (int j = tid; j < M; j += 256) {
    if (j != i && y_cls[j % T] == labi) {
      float l = lrow[j] - rmax;
      psum += l - logf(expf(l) + neg);
      pcnt += 1.f;
    }
  }
  red[tid] = psum; red2[tid] = pcnt; __syncthreads();
  for (int off = 128; off > 0; off >>= 1) {
    if (tid < off) { red[tid] += red[tid + off]; red2[tid] += red2[tid + off]; }
    __syncthreads();
  }
  if (tid == 0) {
    float mlpp = red[0] / (red2[0] + 1e-6f);
    atomicAdd(out, -0.5f * mlpp / (float)M);   // LOSS_WEIGHT * mean(-T/B * mlpp)
  }
}

// ---------------------------------------------------------------------------
extern "C" void kernel_launch(void* const* d_in, const int* in_sizes, int n_in,
                              void* d_out, int out_size, void* d_ws, size_t ws_size,
                              hipStream_t stream) {
  const float* preds = (const float*)d_in[0];
  const float* w1    = (const float*)d_in[1];
  const float* b1    = (const float*)d_in[2];
  const float* gamma = (const float*)d_in[3];
  const float* beta  = (const float*)d_in[4];
  const float* w2    = (const float*)d_in[5];
  const float* b2    = (const float*)d_in[6];
  const int* batch_idx = (const int*)d_in[7];
  const int* pix_idx   = (const int*)d_in[8];
  const int* y_cls     = (const int*)d_in[9];
  float* out = (float*)d_out;
  (void)n_in; (void)out_size; (void)ws_size;

  const int T      = in_sizes[7];
  const int n_view = in_sizes[8] / T;
  const int M      = T * n_view;
  const int Mpad   = ((M + 15) / 16) * 16;
  const int Mt     = Mpad / 16;

  char* ws = (char*)d_ws;
  size_t off = 0;
  auto carve = [&](size_t bytes) -> char* {
    char* p = ws + off;
    off = (off + bytes + 255) & ~(size_t)255;
    return p;
  };
  float* S       = (float*)carve(512 * 512 * sizeof(float));
  float* colsum  = (float*)carve(512 * sizeof(float));
  float* meanA   = (float*)carve(512 * sizeof(float));
  float* invstdA = (float*)carve(512 * sizeof(float));
  unsigned short* Fb = (unsigned short*)carve((size_t)Mpad * FDIM * sizeof(unsigned short));
  float* logits  = (float*)carve((size_t)Mpad * Mpad * sizeof(float));

  {
    int nz = 512 * 512 + 3 * 512;  // S, colsum, mean, invstd are contiguous (256B-multiples)
    k_zero<<<dim3((nz + 255) / 256), dim3(256), 0, stream>>>(S, nz, out);
  }
  k_colsum<<<dim3(512), dim3(256), 0, stream>>>(preds, colsum);
  k_gram<<<dim3(4, 4, 16), dim3(256), 0, stream>>>(preds, S);
  k_stats<<<dim3(512), dim3(256), 0, stream>>>(S, colsum, w1, b1, meanA, invstdA);
  k_embed<<<dim3(Mpad / 16), dim3(256), 0, stream>>>(
      preds, w1, b1, gamma, beta, w2, b2, meanA, invstdA,
      batch_idx, pix_idx, T, n_view, M, Fb);
  k_logits<<<dim3(Mt, Mt), dim3(32), 0, stream>>>(Fb, logits, Mpad);
  k_loss<<<dim3(M), dim3(256), (size_t)Mpad * sizeof(float), stream>>>(
      logits, y_cls, T, M, Mpad, out);
}
